// CUBASpikingCNN_8847632629954
// MI455X (gfx1250) — matrix operand order repaired and checked by reading, not compiled, not run on hardware
//
#include <hip/hip_runtime.h>
#include <hip/hip_bf16.h>

// ---------------------------------------------------------------------------
// CUBA-LIF spiking CNN, T=32, B=256 on gfx1250.
// GEMM layers use v_wmma_f32_16x16x32_f16; each wave computes a 16x64 strip
// (4 accumulators) so one A fragment feeds 4 WMMAs. B fragments are staged
// block-cooperatively into LDS with GLOBAL_LOAD_ASYNC_TO_LDS_B128 (ASYNCcnt,
// double-buffered) and consumed via ds_load. LIF fused into the epilogue.
// ---------------------------------------------------------------------------

typedef __attribute__((ext_vector_type(16))) _Float16 v16h;
typedef __attribute__((ext_vector_type(8)))  _Float16 v8h;
typedef __attribute__((ext_vector_type(8)))  float    v8f;
typedef int v4i_ __attribute__((vector_size(16)));   // matches builtin param pointee

#define BATCH   256
#define TSTEPS  32
#define CDECAY  0.5f
#define VDECAY  0.75f
#define VTH     0.5f

// ---------------------------------------------------------------------------
// async global->LDS copy of 32 bytes per thread (two b128), ASYNCcnt-tracked
// ---------------------------------------------------------------------------
__device__ __forceinline__ void async_copy32(const _Float16* src, _Float16* dst) {
#if __has_builtin(__builtin_amdgcn_global_load_async_to_lds_b128)
    __attribute__((address_space(1))) v4i_* g0 =
        (__attribute__((address_space(1))) v4i_*)(uintptr_t)src;
    __attribute__((address_space(1))) v4i_* g1 =
        (__attribute__((address_space(1))) v4i_*)(uintptr_t)(src + 8);
    __attribute__((address_space(3))) v4i_* l0 =
        (__attribute__((address_space(3))) v4i_*)(unsigned)(uintptr_t)dst;
    __attribute__((address_space(3))) v4i_* l1 =
        (__attribute__((address_space(3))) v4i_*)(unsigned)(uintptr_t)(dst + 8);
    __builtin_amdgcn_global_load_async_to_lds_b128(g0, l0, 0, 0);
    __builtin_amdgcn_global_load_async_to_lds_b128(g1, l1, 0, 0);
#else
    unsigned lo = (unsigned)(uintptr_t)dst;  // LDS byte offset = low 32 bits
    asm volatile("global_load_async_to_lds_b128 %0, %1, off"
                 :: "v"(lo), "v"(src) : "memory");
    asm volatile("global_load_async_to_lds_b128 %0, %1, off"
                 :: "v"(lo + 16u), "v"(src + 8) : "memory");
#endif
}

__device__ __forceinline__ void wait_async_le(int n) {
#if __has_builtin(__builtin_amdgcn_s_wait_asynccnt)
    if (n == 0) __builtin_amdgcn_s_wait_asynccnt(0);
    else        __builtin_amdgcn_s_wait_asynccnt(2);
#else
    if (n == 0) asm volatile("s_wait_asynccnt 0x0" ::: "memory");
    else        asm volatile("s_wait_asynccnt 0x2" ::: "memory");
#endif
}

__device__ __forceinline__ v8f wmma_f16(v16h a, v16h b, v8f c) {
    return __builtin_amdgcn_wmma_f32_16x16x32_f16(false, a, false, b, (short)0, c,
                                                  false, false);
}

// ---------------------------------------------------------------------------
// Utility: zero a word range (grid-stride)
// ---------------------------------------------------------------------------
__global__ void k_zero(unsigned int* p, size_t nwords) {
    size_t i = (size_t)blockIdx.x * blockDim.x + threadIdx.x;
    size_t stride = (size_t)gridDim.x * blockDim.x;
    for (; i < nwords; i += stride) p[i] = 0u;
}

// ---------------------------------------------------------------------------
// Weight packing into WMMA B-fragment order: element (k,n) of [K x N]:
//   lane = (n%16) + 16*(((k%32)/8)%2), idx = (k%32)%8 + 8*((k%32)/16)
//   dst[(( (k/32)*ntiles + n/16 )*32 + lane)*16 + idx]
// ---------------------------------------------------------------------------
__device__ __forceinline__ size_t bfrag_off(int k, int n, int N) {
    int kt = k >> 5, kk = k & 31, nt = n >> 4;
    int lane = (n & 15) + (((kk >> 3) & 1) << 4);
    int idx = (kk & 7) + ((kk >> 4) << 3);
    int ntiles = N >> 4;
    return (((size_t)kt * ntiles + nt) * 32 + lane) * 16 + idx;
}

__global__ void k_pack_dense(const float* __restrict__ src, _Float16* __restrict__ dst,
                             int K, int N) {
    int tid = blockIdx.x * blockDim.x + threadIdx.x;
    if (tid >= K * N) return;
    int n = tid / K, k = tid % K;
    dst[bfrag_off(k, n, N)] = (_Float16)src[(size_t)n * K + k];
}

__global__ void k_pack_conv(const float* __restrict__ src, _Float16* __restrict__ dst,
                            int CI, int N) {
    int tot = N * CI * 9;
    int tid = blockIdx.x * blockDim.x + threadIdx.x;
    if (tid >= tot) return;
    int n = tid / (CI * 9);
    int r = tid % (CI * 9);
    int ci = r / 9, p = r % 9;
    int k = p * CI + ci;
    dst[bfrag_off(k, n, 9 * CI)] = (_Float16)src[((size_t)n * CI + ci) * 9 + p];
}

// ---------------------------------------------------------------------------
// Fused LIF epilogue on a D-fragment: element r -> (m = mbase + r, n = col)
// ---------------------------------------------------------------------------
__device__ __forceinline__ void lif_store(v8f acc, float bias,
                                          const _Float16* __restrict__ addend,
                                          int mbase, int col, int N,
                                          float* __restrict__ cur,
                                          float* __restrict__ vlt,
                                          const _Float16* __restrict__ spk_prev,
                                          _Float16* __restrict__ spk_out) {
#pragma unroll
    for (int r = 0; r < 8; ++r) {
        int m = mbase + r;
        size_t idx = (size_t)m * N + col;
        float z = acc[r] + bias;
        if (addend) z += (float)addend[idx];
        float c = CDECAY * cur[idx] + z;
        float sp_prev = (float)spk_prev[idx];
        float v = VDECAY * vlt[idx] * (1.0f - sp_prev) + c;
        float sp = (v > VTH) ? 1.0f : 0.0f;
        cur[idx] = c;
        vlt[idx] = v;
        spk_out[idx] = (_Float16)sp;
    }
}

// Build a 16-f16 A fragment from two contiguous 8-f16 runs (ISA A layout:
// lane group 0 holds K g*8..g*8+7 and 16+g*8.. ; g = lane/16).
__device__ __forceinline__ v16h make_a(const _Float16* p) {
    v8h lo = *(const v8h*)p;
    v8h hi = *(const v8h*)(p + 16);
    v16h a;
#pragma unroll
    for (int i = 0; i < 8; ++i) { a[i] = lo[i]; a[8 + i] = hi[i]; }
    return a;
}

// Read one B fragment (lane-major, 16 f16/lane) from staged LDS tile
__device__ __forceinline__ v16h lds_b(const _Float16* bb, int j, int lane) {
    return *(const v16h*)(bb + ((size_t)(j * 32 + lane)) * 16);
}

// ---------------------------------------------------------------------------
// conv1 (K=9, too skinny for WMMA): direct f32 conv + LIF.
// Output channels-last: c1spk[b][oy][ox][oc], 8x8x64 per image.
// ---------------------------------------------------------------------------
__global__ void k_conv1_lif(const float* __restrict__ x, const float* __restrict__ w,
                            const float* __restrict__ bias, int t,
                            float* __restrict__ cur, float* __restrict__ vlt,
                            _Float16* __restrict__ spk) {
    int tid = blockIdx.x * blockDim.x + threadIdx.x;
    if (tid >= BATCH * 8 * 8 * 64) return;
    int oc = tid & 63;
    int s = tid >> 6;
    int ox = s & 7; s >>= 3;
    int oy = s & 7; s >>= 3;
    int b = s;
    float z = bias[oc];
#pragma unroll
    for (int kh = 0; kh < 3; ++kh)
#pragma unroll
        for (int kw = 0; kw < 3; ++kw) {
            float xv = x[((size_t)(b * 100 + (oy + kh) * 10 + (ox + kw))) * TSTEPS + t];
            z += xv * w[oc * 9 + kh * 3 + kw];
        }
    float c = CDECAY * cur[tid] + z;
    float sp_prev = (float)spk[tid];
    float v = VDECAY * vlt[tid] * (1.0f - sp_prev) + c;
    float sp = (v > VTH) ? 1.0f : 0.0f;
    cur[tid] = c; vlt[tid] = v; spk[tid] = (_Float16)sp;
}

// ---------------------------------------------------------------------------
// conv2 implicit-GEMM: M = B*36 (b,oy,ox), N = 128, K = 576 (18 k-slabs).
// Block = 4 waves = 4 m-tiles x one 64-col n-group; B slab (4KB) staged in LDS.
// grid(144, 2), block(32,4).
// ---------------------------------------------------------------------------
__global__ void k_conv2_wmma(const _Float16* __restrict__ c1, const _Float16* __restrict__ Wp,
                             const float* __restrict__ bias,
                             float* __restrict__ cur, float* __restrict__ vlt,
                             _Float16* __restrict__ spk) {
    const int N = 128, ntiles = N >> 4, KT = 18;
    __shared__ _Float16 bsh[2][2048];
    int lane = threadIdx.x;
    int tl = threadIdx.y * 32 + lane;
    int ng = blockIdx.y;
    int mt = blockIdx.x * 4 + threadIdx.y;
    int g = lane >> 4;
    int mrow = mt * 16 + (lane & 15);
    int b = mrow / 36, rem = mrow % 36;
    int oy = rem / 6, ox = rem % 6;
    const _Float16* base = c1 + (size_t)b * 4096;
    v8f acc[4] = {};
    async_copy32(Wp + ((size_t)0 * ntiles + ng * 4) * 512 + (size_t)tl * 16,
                 &bsh[0][tl * 16]);
    for (int kt = 0; kt < KT; ++kt) {
        int buf = kt & 1;
        if (kt + 1 < KT) {
            async_copy32(Wp + ((size_t)(kt + 1) * ntiles + ng * 4) * 512 + (size_t)tl * 16,
                         &bsh[buf ^ 1][tl * 16]);
            wait_async_le(2);
        } else {
            wait_async_le(0);
        }
        __syncthreads();
        int p = kt >> 1, cib = (kt & 1) << 5;
        int kh = p / 3, kw = p % 3;
        const _Float16* pa = base + ((oy + kh) * 8 + (ox + kw)) * 64 + cib + g * 8;
        v16h a = make_a(pa);
        const _Float16* bb = bsh[buf];
        acc[0] = wmma_f16(a, lds_b(bb, 0, lane), acc[0]);
        acc[1] = wmma_f16(a, lds_b(bb, 1, lane), acc[1]);
        acc[2] = wmma_f16(a, lds_b(bb, 2, lane), acc[2]);
        acc[3] = wmma_f16(a, lds_b(bb, 3, lane), acc[3]);
        __syncthreads();
    }
    int mbase = mt * 16 + 8 * g;
#pragma unroll
    for (int j = 0; j < 4; ++j) {
        int col = (ng * 4 + j) * 16 + (lane & 15);
        lif_store(acc[j], bias[col], nullptr, mbase, col, N, cur, vlt, spk, spk);
    }
}

// ---------------------------------------------------------------------------
// AvgPool2d(2): [B][6][6][128] -> [B][3][3][128] (channels-last)
// ---------------------------------------------------------------------------
__global__ void k_pool(const _Float16* __restrict__ c2, _Float16* __restrict__ out) {
    int tid = blockIdx.x * blockDim.x + threadIdx.x;
    if (tid >= BATCH * 3 * 3 * 128) return;
    int c = tid & 127;
    int s = tid >> 7;
    int px = s % 3; s /= 3;
    int py = s % 3; s /= 3;
    int b = s;
    float sum = 0.f;
#pragma unroll
    for (int dy = 0; dy < 2; ++dy)
#pragma unroll
        for (int dx = 0; dx < 2; ++dx)
            sum += (float)c2[(((size_t)b * 6 + (2 * py + dy)) * 6 + (2 * px + dx)) * 128 + c];
    out[tid] = (_Float16)(sum * 0.25f);
}

// ---------------------------------------------------------------------------
// conv3 implicit-GEMM: M = B, N = 256, K = 1152 (36 slabs). grid(4,4).
// LIF reset reads previous temporal frame; spikes write current frame.
// ---------------------------------------------------------------------------
__global__ void k_conv3_wmma(const _Float16* __restrict__ pooled, const _Float16* __restrict__ Wp,
                             const float* __restrict__ bias,
                             float* __restrict__ cur, float* __restrict__ vlt,
                             const _Float16* __restrict__ spk_prev,
                             _Float16* __restrict__ spk_out) {
    const int N = 256, ntiles = N >> 4, KT = 36;
    __shared__ _Float16 bsh[2][2048];
    int lane = threadIdx.x;
    int tl = threadIdx.y * 32 + lane;
    int ng = blockIdx.y;
    int mt = blockIdx.x * 4 + threadIdx.y;
    int g = lane >> 4;
    int b = mt * 16 + (lane & 15);
    const _Float16* base = pooled + (size_t)b * 1152;
    v8f acc[4] = {};
    async_copy32(Wp + ((size_t)0 * ntiles + ng * 4) * 512 + (size_t)tl * 16,
                 &bsh[0][tl * 16]);
    for (int kt = 0; kt < KT; ++kt) {
        int buf = kt & 1;
        if (kt + 1 < KT) {
            async_copy32(Wp + ((size_t)(kt + 1) * ntiles + ng * 4) * 512 + (size_t)tl * 16,
                         &bsh[buf ^ 1][tl * 16]);
            wait_async_le(2);
        } else {
            wait_async_le(0);
        }
        __syncthreads();
        int p = kt >> 2, cib = (kt & 3) << 5;
        const _Float16* pa = base + p * 128 + cib + g * 8;
        v16h a = make_a(pa);
        const _Float16* bb = bsh[buf];
        acc[0] = wmma_f16(a, lds_b(bb, 0, lane), acc[0]);
        acc[1] = wmma_f16(a, lds_b(bb, 1, lane), acc[1]);
        acc[2] = wmma_f16(a, lds_b(bb, 2, lane), acc[2]);
        acc[3] = wmma_f16(a, lds_b(bb, 3, lane), acc[3]);
        __syncthreads();
    }
    int mbase = mt * 16 + 8 * g;
#pragma unroll
    for (int j = 0; j < 4; ++j) {
        int col = (ng * 4 + j) * 16 + (lane & 15);
        lif_store(acc[j], bias[col], nullptr, mbase, col, N, cur, vlt, spk_prev, spk_out);
    }
}

// ---------------------------------------------------------------------------
// Dense GEMM (+ up to 3 taps) + optional addend + fused LIF.
// A: f16 row-major [M][K]; W: packed fragments. grid(M/64, N/64), block(32,4).
// ---------------------------------------------------------------------------
__global__ void k_dense_wmma(const _Float16* __restrict__ A0, const _Float16* __restrict__ A1,
                             const _Float16* __restrict__ A2,
                             const _Float16* __restrict__ W0, const _Float16* __restrict__ W1,
                             const _Float16* __restrict__ W2,
                             const float* __restrict__ b0, const float* __restrict__ b1,
                             const float* __restrict__ b2,
                             int ntaps, const _Float16* __restrict__ addend,
                             int N, int K,
                             float* __restrict__ cur, float* __restrict__ vlt,
                             const _Float16* __restrict__ spk_prev,
                             _Float16* __restrict__ spk_out) {
    __shared__ _Float16 bsh[2][2048];
    int ntiles = N >> 4;
    int lane = threadIdx.x;
    int tl = threadIdx.y * 32 + lane;
    int ng = blockIdx.y;
    int mt = blockIdx.x * 4 + threadIdx.y;
    int g = lane >> 4;
    int m = mt * 16 + (lane & 15);
    const _Float16* As[3] = {A0, A1, A2};
    const _Float16* Ws[3] = {W0, W1, W2};
    int kpt = K >> 5;            // k-slabs per tap
    int QT = ntaps * kpt;
    v8f acc[4] = {};
    async_copy32(Ws[0] + ((size_t)0 * ntiles + ng * 4) * 512 + (size_t)tl * 16,
                 &bsh[0][tl * 16]);
    for (int q = 0; q < QT; ++q) {
        int buf = q & 1;
        if (q + 1 < QT) {
            int qq = q + 1;
            async_copy32(Ws[qq / kpt] + ((size_t)(qq % kpt) * ntiles + ng * 4) * 512 +
                             (size_t)tl * 16,
                         &bsh[buf ^ 1][tl * 16]);
            wait_async_le(2);
        } else {
            wait_async_le(0);
        }
        __syncthreads();
        int tap = q / kpt, kt = q % kpt;
        const _Float16* pa = As[tap] + (size_t)m * K + kt * 32 + g * 8;
        __builtin_prefetch(pa + 32, 0, 3);   // next k-slab of this A row
        v16h a = make_a(pa);
        const _Float16* bb = bsh[buf];
        acc[0] = wmma_f16(a, lds_b(bb, 0, lane), acc[0]);
        acc[1] = wmma_f16(a, lds_b(bb, 1, lane), acc[1]);
        acc[2] = wmma_f16(a, lds_b(bb, 2, lane), acc[2]);
        acc[3] = wmma_f16(a, lds_b(bb, 3, lane), acc[3]);
        __syncthreads();
    }
    int mbase = mt * 16 + 8 * g;
#pragma unroll
    for (int j = 0; j < 4; ++j) {
        int col = (ng * 4 + j) * 16 + (lane & 15);
        float bias = 0.f;
        if (b0) bias += b0[col];
        if (ntaps > 1 && b1) bias += b1[col];
        if (ntaps > 2 && b2) bias += b2[col];
        lif_store(acc[j], bias, addend, mbase, col, N, cur, vlt, spk_prev, spk_out);
    }
}

// ---------------------------------------------------------------------------
// Readout: acc[b][j] += ts[t] * dot(f1spk[b,:], fc2_w[j,:])
// ---------------------------------------------------------------------------
__global__ void k_fc2_acc(const _Float16* __restrict__ f1, const float* __restrict__ w,
                          const float* __restrict__ ts, int t, float* __restrict__ out) {
    int tid = blockIdx.x * blockDim.x + threadIdx.x;
    if (tid >= BATCH * 2) return;
    int b = tid >> 1, j = tid & 1;
    float s = 0.f;
#pragma unroll 4
    for (int i = 0; i < 128; ++i)
        s += (float)f1[b * 128 + i] * w[j * 128 + i];
    out[tid] += ts[t] * s;
}

// ---------------------------------------------------------------------------
// Host launcher
// ---------------------------------------------------------------------------
extern "C" void kernel_launch(void* const* d_in, const int* in_sizes, int n_in,
                              void* d_out, int out_size, void* d_ws, size_t ws_size,
                              hipStream_t stream) {
    const float* in_x = (const float*)d_in[0];
    const float* w1   = (const float*)d_in[1];
    const float* b1   = (const float*)d_in[2];
    const float* w2   = (const float*)d_in[3];
    const float* b2   = (const float*)d_in[4];
    const float* w3   = (const float*)d_in[5];
    const float* b3   = (const float*)d_in[6];
    const float* tcw  = (const float*)d_in[7];
    const float* tcb  = (const float*)d_in[8];
    const float* recw = (const float*)d_in[9];
    const float* recb = (const float*)d_in[10];
    const float* f1w  = (const float*)d_in[11];
    const float* f1b  = (const float*)d_in[12];
    const float* f2w  = (const float*)d_in[13];
    const float* tsw  = (const float*)d_in[14];
    float* out = (float*)d_out;
    char* ws = (char*)d_ws;

    auto al = [](size_t x) { return (x + 255) & ~(size_t)255; };
    size_t off = 0;
    auto take = [&](size_t bytes) { size_t o = off; off = al(off + bytes); return o; };

    size_t o_w2p  = take(576 * 128 * 2);
    size_t o_w3p  = take(1152 * 256 * 2);
    size_t o_wtc  = take(3 * 256 * 256 * 2);
    size_t o_wrec = take(256 * 256 * 2);
    size_t o_wfc1 = take(256 * 128 * 2);
    size_t zero_start = off;
    size_t o_c1s  = take((size_t)BATCH * 64 * 64 * 2);
    size_t o_c2s  = take((size_t)BATCH * 36 * 128 * 2);
    size_t o_pool = take((size_t)BATCH * 9 * 128 * 2);
    size_t o_fr   = take((size_t)3 * BATCH * 256 * 2);
    size_t o_tcs  = take((size_t)BATCH * 256 * 2);
    size_t o_rs0  = take((size_t)BATCH * 256 * 2);
    size_t o_rs1  = take((size_t)BATCH * 256 * 2);
    size_t o_f1s  = take((size_t)BATCH * 128 * 2);
    size_t o_c1c = take((size_t)BATCH * 4096 * 4);
    size_t o_c1v = take((size_t)BATCH * 4096 * 4);
    size_t o_c2c = take((size_t)BATCH * 4608 * 4);
    size_t o_c2v = take((size_t)BATCH * 4608 * 4);
    size_t o_c3c = take((size_t)BATCH * 256 * 4);
    size_t o_c3v = take((size_t)BATCH * 256 * 4);
    size_t o_tcc = take((size_t)BATCH * 256 * 4);
    size_t o_tcv = take((size_t)BATCH * 256 * 4);
    size_t o_rc  = take((size_t)BATCH * 256 * 4);
    size_t o_rv  = take((size_t)BATCH * 256 * 4);
    size_t o_f1c = take((size_t)BATCH * 128 * 4);
    size_t o_f1v = take((size_t)BATCH * 128 * 4);
    size_t total = off;
    (void)total; (void)ws_size; (void)in_sizes; (void)n_in; (void)out_size;

    _Float16* w2p  = (_Float16*)(ws + o_w2p);
    _Float16* w3p  = (_Float16*)(ws + o_w3p);
    _Float16* wtcp = (_Float16*)(ws + o_wtc);
    _Float16* wrecp= (_Float16*)(ws + o_wrec);
    _Float16* wfc1p= (_Float16*)(ws + o_wfc1);
    _Float16* c1s  = (_Float16*)(ws + o_c1s);
    _Float16* c2s  = (_Float16*)(ws + o_c2s);
    _Float16* pool = (_Float16*)(ws + o_pool);
    _Float16* fr   = (_Float16*)(ws + o_fr);
    _Float16* tcs  = (_Float16*)(ws + o_tcs);
    _Float16* rs[2] = {(_Float16*)(ws + o_rs0), (_Float16*)(ws + o_rs1)};
    _Float16* f1s  = (_Float16*)(ws + o_f1s);
    float* c1c = (float*)(ws + o_c1c); float* c1v = (float*)(ws + o_c1v);
    float* c2c = (float*)(ws + o_c2c); float* c2v = (float*)(ws + o_c2v);
    float* c3c = (float*)(ws + o_c3c); float* c3v = (float*)(ws + o_c3v);
    float* tcc = (float*)(ws + o_tcc); float* tcv = (float*)(ws + o_tcv);
    float* rc  = (float*)(ws + o_rc);  float* rv  = (float*)(ws + o_rv);
    float* f1c = (float*)(ws + o_f1c); float* f1v = (float*)(ws + o_f1v);

    const size_t FRAME = (size_t)BATCH * 256;

    size_t zwords = (total - zero_start) / 4;
    k_zero<<<2048, 256, 0, stream>>>((unsigned int*)(ws + zero_start), zwords);
    k_zero<<<2, 256, 0, stream>>>((unsigned int*)out, (size_t)BATCH * 2);

    k_pack_conv<<<(128 * 64 * 9 + 255) / 256, 256, 0, stream>>>(w2, w2p, 64, 128);
    k_pack_conv<<<(256 * 128 * 9 + 255) / 256, 256, 0, stream>>>(w3, w3p, 128, 256);
    for (int j = 0; j < 3; ++j)
        k_pack_dense<<<256, 256, 0, stream>>>(tcw + (size_t)j * 65536, wtcp + (size_t)j * 65536,
                                              256, 256);
    k_pack_dense<<<256, 256, 0, stream>>>(recw, wrecp, 256, 256);
    k_pack_dense<<<128, 256, 0, stream>>>(f1w, wfc1p, 256, 128);

    dim3 wblk(32, 4);
    for (int t = 0; t < TSTEPS; ++t) {
        k_conv1_lif<<<(BATCH * 64 * 64 + 255) / 256, 256, 0, stream>>>(
            in_x, w1, b1, t, c1c, c1v, c1s);
        k_conv2_wmma<<<dim3(144, 2), wblk, 0, stream>>>(c1s, w2p, b2, c2c, c2v, c2s);
        k_pool<<<(BATCH * 9 * 128 + 255) / 256, 256, 0, stream>>>(c2s, pool);
        _Float16* fr_out  = fr + (size_t)(t % 3) * FRAME;
        _Float16* fr_prev = fr + (size_t)((t + 2) % 3) * FRAME;
        k_conv3_wmma<<<dim3(4, 4), wblk, 0, stream>>>(pool, w3p, b3, c3c, c3v, fr_prev, fr_out);
        int len = (t + 1 < 3) ? (t + 1) : 3;
        const _Float16* A[3] = {nullptr, nullptr, nullptr};
        const _Float16* W[3] = {nullptr, nullptr, nullptr};
        const float* Bv[3] = {nullptr, nullptr, nullptr};
        for (int j = 0; j < len; ++j) {
            int s = t - len + 1 + j;
            A[j] = fr + (size_t)(s % 3) * FRAME;
            W[j] = wtcp + (size_t)j * 65536;
            Bv[j] = tcb + (size_t)j * 256;
        }
        k_dense_wmma<<<dim3(4, 4), wblk, 0, stream>>>(
            A[0], A[1], A[2], W[0], W[1], W[2], Bv[0], Bv[1], Bv[2],
            len, nullptr, 256, 256, tcc, tcv, tcs, tcs);
        _Float16* r_prev = rs[t & 1];
        _Float16* r_cur  = rs[(t + 1) & 1];
        k_dense_wmma<<<dim3(4, 4), wblk, 0, stream>>>(
            r_prev, nullptr, nullptr, wrecp, nullptr, nullptr, recb, nullptr, nullptr,
            1, tcs, 256, 256, rc, rv, r_prev, r_cur);
        k_dense_wmma<<<dim3(4, 2), wblk, 0, stream>>>(
            r_cur, nullptr, nullptr, wfc1p, nullptr, nullptr, f1b, nullptr, nullptr,
            1, nullptr, 128, 256, f1c, f1v, f1s, f1s);
        k_fc2_acc<<<2, 256, 0, stream>>>(f1s, f2w, tsw, t, out);
    }
}
